// MultiHeadAttention_26285199852234
// MI455X (gfx1250) — matrix operand order repaired and checked by reading, not compiled, not run on hardware
//
#include <hip/hip_runtime.h>

// ---------------- problem constants ----------------
constexpr int Bz = 4;
constexpr int T  = 2048;
constexpr int E  = 512;
constexpr int H  = 8;
constexpr int Dh = 64;
constexpr int HD = H * Dh; // 512

typedef __attribute__((ext_vector_type(16))) __bf16 v16bf;
typedef __attribute__((ext_vector_type(8)))  float  v8f;
typedef __attribute__((ext_vector_type(4)))  unsigned int v4u;
typedef __attribute__((ext_vector_type(8)))  int  v8i;
typedef __attribute__((ext_vector_type(4)))  int  v4i;

#define WMMA_BF16(a, b, c) \
  __builtin_amdgcn_wmma_f32_16x16x32_bf16(false, (a), false, (b), (short)0, (c), false, false)

// ---------------- fragment loaders (wave32 layouts per CDNA5 ISA) --------
// A matrix 16x32 bf16: lane<16 -> row m=lane, K = {0..7, 16..23}; lane>=16 -> row m,
// K = {8..15, 24..31}.  Two contiguous 16B chunks per lane.
__device__ __forceinline__ v16bf load_fragA(const __bf16* base, int ld, int k0) {
  int lane = threadIdx.x & 31;
  int m  = lane & 15;
  int kb = (lane >> 4) << 3; // 0 or 8
  const __bf16* p = base + (size_t)m * ld + (k0 + kb);
  union { v16bf v; float4 f[2]; } u;
  u.f[0] = *(const float4*)(p);
  u.f[1] = *(const float4*)(p + 16);
  return u.v;
}

// B matrix 32x16 bf16: lane<16 -> col n=lane, K = {0..7, 16..23}; lane>=16 -> col n,
// K = {8..15, 24..31}.  Two contiguous 16B chunks per lane; base points at the K-major matrix (row n = column n).
__device__ __forceinline__ v16bf load_fragB(const __bf16* base, int ld, int k0) {
  int lane = threadIdx.x & 31;
  int n  = lane & 15;
  int kb = (lane >> 4) << 3; // 0 or 8
  const __bf16* p = base + (size_t)n * ld + (k0 + kb);
  union { v16bf v; float4 f[2]; } u;
  u.f[0] = *(const float4*)(p);
  u.f[1] = *(const float4*)(p + 16);
  return u.v;
}

// ---------------- Tensor Data Mover: 2D tile (rows x cols bf16) global -> LDS --------
// D# layout per CDNA5 ISA.  lds_off is a byte offset into
// the workgroup's LDS allocation; tile is stored packed (row r at lds_off + r*cols*2).
__device__ __forceinline__ void tdm_load_2d(unsigned lds_off, const __bf16* gptr,
                                            int cols, int rows, int row_stride) {
  unsigned long long ga = (unsigned long long)(uintptr_t)gptr;
  v4u g0;
  g0[0] = 1u;                                   // count=1, is_restore=0, gather off
  g0[1] = lds_off;                              // lds_addr (bytes)
  g0[2] = (unsigned)(ga & 0xffffffffu);         // global_addr[31:0]
  g0[3] = (unsigned)((ga >> 32) & 0x1ffffffu)   // global_addr[56:32]
        | (2u << 30);                           // type=2 ("image")
  v8i g1;
  g1[0] = (1 << 16);                            // workgroup_mask=0, data_size=1 (2 bytes)
  g1[1] = (cols & 0xffff) << 16;                // tensor_dim0[15:0] = cols
  g1[2] = ((unsigned)cols >> 16)                // tensor_dim0[31:16]
        | ((rows & 0xffff) << 16);              // tensor_dim1[15:0] = rows
  g1[3] = ((unsigned)rows >> 16)                // tensor_dim1[31:16]
        | ((cols & 0xffff) << 16);              // tile_dim0 = cols
  g1[4] = (rows & 0xffff);                      // tile_dim1 = rows, tile_dim2 = 0
  g1[5] = row_stride;                           // tensor_dim0_stride[31:0]
  g1[6] = 0;                                    // stride hi / dim1_stride lo
  g1[7] = 0;
  v4i g2 = {0, 0, 0, 0};
  v4i g3 = {0, 0, 0, 0};
#if defined(__clang_major__) && (__clang_major__ >= 23)
  v8i g4 = {0, 0, 0, 0, 0, 0, 0, 0};
  __builtin_amdgcn_tensor_load_to_lds(g0, g1, g2, g3, g4, 0);
#else
  __builtin_amdgcn_tensor_load_to_lds(g0, g1, g2, g3, 0);
#endif
}

// ---------------- conversion / transpose kernels ----------------
__global__ void cvt_bf16_kernel(const float* __restrict__ in, __bf16* __restrict__ out, int n) {
  for (int i = blockIdx.x * blockDim.x + threadIdx.x; i < n; i += gridDim.x * blockDim.x)
    out[i] = (__bf16)in[i];
}

// batched transpose: in[batch][R][C] (f32) -> out[batch][C][R] (bf16); batch = blockIdx.y
__global__ void transpose_bf16_kernel(const float* __restrict__ in, __bf16* __restrict__ out,
                                      int R, int C) {
  const float* ib = in  + (size_t)blockIdx.y * R * C;
  __bf16*      ob = out + (size_t)blockIdx.y * R * C;
  int n = R * C;
  for (int i = blockIdx.x * blockDim.x + threadIdx.x; i < n; i += gridDim.x * blockDim.x) {
    int r = i / C, c = i % C;
    ob[(size_t)c * R + r] = (__bf16)ib[i];
  }
}

// ---------------- fused QKV projection: per (b,h) GEMM [T x E] * [E x 64] -----------
// One wave computes a 32x64 strip of Q, K and V: two A-fragments so every B-fragment
// is reused twice (2 A-loads + 12 B-loads -> 24 WMMAs per k-step).
__global__ void __launch_bounds__(128)
qkv_kernel(const __bf16* __restrict__ xb,
           const __bf16* __restrict__ Wqt, const __bf16* __restrict__ Wkt,
           const __bf16* __restrict__ Wvt,
           const float* __restrict__ bq, const float* __restrict__ bk,
           const float* __restrict__ bv,
           __bf16* __restrict__ Qb, __bf16* __restrict__ Kb, __bf16* __restrict__ Vtb) {
  int wid  = threadIdx.x >> 5;
  int lane = threadIdx.x & 31;
  int g  = blockIdx.x * 4 + wid;       // 2048 wave-tiles total
  int mt = g & 63;                     // T/32 tiles
  int h  = (g >> 6) & 7;
  int b  = g >> 9;
  int row0 = mt * 32;

  const __bf16* xrow = xb + ((size_t)(b * T + row0)) * E;
  const __bf16* wq = Wqt + (size_t)h * Dh * E;
  const __bf16* wk = Wkt + (size_t)h * Dh * E;
  const __bf16* wv = Wvt + (size_t)h * Dh * E;

  v8f qa[2][4] = {};
  v8f ka[2][4] = {};
  v8f va[2][4] = {};

  for (int k0 = 0; k0 < E; k0 += 32) {
    v16bf a0 = load_fragA(xrow,               E, k0);
    v16bf a1 = load_fragA(xrow + 16 * (size_t)E, E, k0);
#pragma unroll
    for (int n = 0; n < 4; ++n) {
      v16bf bqf = load_fragB(wq + (size_t)(n * 16) * E, E, k0);
      qa[0][n] = WMMA_BF16(a0, bqf, qa[0][n]);
      qa[1][n] = WMMA_BF16(a1, bqf, qa[1][n]);
      v16bf bkf = load_fragB(wk + (size_t)(n * 16) * E, E, k0);
      ka[0][n] = WMMA_BF16(a0, bkf, ka[0][n]);
      ka[1][n] = WMMA_BF16(a1, bkf, ka[1][n]);
      v16bf bvf = load_fragB(wv + (size_t)(n * 16) * E, E, k0);
      va[0][n] = WMMA_BF16(a0, bvf, va[0][n]);
      va[1][n] = WMMA_BF16(a1, bvf, va[1][n]);
    }
  }

  int mb = (lane >> 4) << 3;  // row base in tile (0 or 8)
  int nn = lane & 15;         // column within tile
  size_t bh = (size_t)(b * H + h);
  __bf16* Vt = Vtb + bh * Dh * T;

#pragma unroll
  for (int m = 0; m < 2; ++m) {
    int r0 = row0 + m * 16;
    __bf16* Qrow = Qb + bh * T * Dh + (size_t)r0 * Dh;
    __bf16* Krow = Kb + bh * T * Dh + (size_t)r0 * Dh;
#pragma unroll
    for (int n = 0; n < 4; ++n) {
      int d = n * 16 + nn;
      float biq = bq[h * Dh + d];
      float bik = bk[h * Dh + d];
      float biv = bv[h * Dh + d];
#pragma unroll
      for (int i = 0; i < 8; ++i) {
        Qrow[(size_t)(mb + i) * Dh + d] = (__bf16)(qa[m][n][i] + biq);
        Krow[(size_t)(mb + i) * Dh + d] = (__bf16)(ka[m][n][i] + bik);
      }
      union { __bf16 hv[8]; float4 f; } u;
#pragma unroll
      for (int i = 0; i < 8; ++i) u.hv[i] = (__bf16)(va[m][n][i] + biv);
      *(float4*)&Vt[(size_t)d * T + r0 + mb] = u.f;  // contiguous transposed store
    }
  }
}

// ---------------- flash attention with TDM-staged K/V in LDS -------------------------
// Block = 8 waves, all on the same (b,h), query rows [q0, q0+128).  Wave 0 drives a
// double-buffered Tensor-DMA pipeline staging K (32x64) and Vt (64x32) tiles into LDS;
// all waves consume them as WMMA B-fragments via ds_load_b128.
// LDS arena (bf16 elements): K tiles @ 0 / 2048, Vt tiles @ 4096 / 6144,
// per-wave P staging @ 8192 + wid*512.  Total 12288 elems = 24 KB.
__global__ void __launch_bounds__(256)
attn_kernel(const __bf16* __restrict__ Qb, const __bf16* __restrict__ Kb,
            const __bf16* __restrict__ Vtb, __bf16* __restrict__ Oc) {
  __shared__ __bf16 smem[12288];
  int wid  = threadIdx.x >> 5;
  int lane = threadIdx.x & 31;
  int qblk = blockIdx.x & 15;          // 16 q-blocks of 128 rows
  int h    = (blockIdx.x >> 4) & 7;
  int b    = blockIdx.x >> 7;
  int q0    = qblk * 128;
  int qbase = q0 + wid * 16;

  const __bf16* Qbh = Qb  + ((size_t)(b * H + h)) * T * Dh;
  const __bf16* Kbh = Kb  + ((size_t)(b * H + h)) * T * Dh;
  const __bf16* Vbh = Vtb + ((size_t)(b * H + h)) * Dh * T;

  v16bf qf0 = load_fragA(Qbh + (size_t)qbase * Dh, Dh, 0);
  v16bf qf1 = load_fragA(Qbh + (size_t)qbase * Dh, Dh, 32);

  v8f o0 = {}, o1 = {}, o2 = {}, o3 = {};
  float mrow[8], lrow[8];
#pragma unroll
  for (int i = 0; i < 8; ++i) { mrow[i] = -3.0e38f; lrow[i] = 0.0f; }

  int mb = (lane >> 4) << 3;
  int nn = lane & 15;
  const float scale = 0.125f; // 1/sqrt(64)
  const int   send  = qbase + 16;  // this wave's causal bound (exclusive)
  const int   smax  = q0 + 128;    // uniform block-level bound

  // prologue: stage first K/V block into buffer 0
  if (wid == 0) {
    tdm_load_2d(/*lds_off=*/0,    Kbh, /*cols=*/Dh, /*rows=*/32, /*stride=*/Dh);
    tdm_load_2d(/*lds_off=*/8192, Vbh, /*cols=*/32, /*rows=*/Dh, /*stride=*/T);
  }

  int bufsel = 0;
  for (int s0 = 0; s0 < smax; s0 += 32) {
    bool more = (s0 + 32) < smax;
    if (wid == 0) {
      if (more) {
        int nb = bufsel ^ 1;
        tdm_load_2d(/*lds_off=*/(unsigned)(nb * 4096),
                    Kbh + (size_t)(s0 + 32) * Dh, Dh, 32, Dh);
        tdm_load_2d(/*lds_off=*/(unsigned)(8192 + nb * 4096),
                    Vbh + (s0 + 32), 32, Dh, T);
        __builtin_amdgcn_s_wait_tensorcnt(2);  // current pair complete (in-order)
      } else {
        __builtin_amdgcn_s_wait_tensorcnt(0);
      }
    }
    __syncthreads();   // K/V tiles for this iteration visible to all waves

    if (s0 < send) {
      const __bf16* kt = &smem[bufsel * 2048];        // 32 x 64, ld=64
      const __bf16* vt = &smem[4096 + bufsel * 2048]; // 64 x 32, ld=32

      // ---- S = Q * K^T for two 16-wide key tiles ----
      v8f s_0 = {}, s_1 = {};
      s_0 = WMMA_BF16(qf0, load_fragB(kt,            Dh, 0),  s_0);
      s_0 = WMMA_BF16(qf1, load_fragB(kt,            Dh, 32), s_0);
      s_1 = WMMA_BF16(qf0, load_fragB(kt + 16 * Dh,  Dh, 0),  s_1);
      s_1 = WMMA_BF16(qf1, load_fragB(kt + 16 * Dh,  Dh, 32), s_1);

      // ---- causal mask + online softmax ----
      int ts0 = s0 + nn, ts1 = ts0 + 16;
      float p0[8], p1[8], alpha[8];
#pragma unroll
      for (int i = 0; i < 8; ++i) {
        int tq = qbase + mb + i;
        p0[i] = (ts0 <= tq) ? s_0[i] * scale : -3.0e38f;
        p1[i] = (ts1 <= tq) ? s_1[i] * scale : -3.0e38f;
      }
#pragma unroll
      for (int i = 0; i < 8; ++i) {
        float r = fmaxf(p0[i], p1[i]);
        for (int msk = 1; msk < 16; msk <<= 1) r = fmaxf(r, __shfl_xor(r, msk, 32));
        float mo = mrow[i];
        float mn = fmaxf(mo, r);
        float al = __expf(mo - mn);
        float e0 = __expf(p0[i] - mn);
        float e1 = __expf(p1[i] - mn);
        p0[i] = e0; p1[i] = e1;
        float rs = e0 + e1;
        for (int msk = 1; msk < 16; msk <<= 1) rs += __shfl_xor(rs, msk, 32);
        mrow[i]  = mn;
        lrow[i]  = lrow[i] * al + rs;
        alpha[i] = al;
      }

      // ---- rescale O; stage P (C-layout) to LDS; reload as A-fragment ----
#pragma unroll
      for (int i = 0; i < 8; ++i) {
        o0[i] *= alpha[i]; o1[i] *= alpha[i]; o2[i] *= alpha[i]; o3[i] *= alpha[i];
        smem[8192 + wid * 512 + (mb + i) * 32 + nn]      = (__bf16)p0[i];
        smem[8192 + wid * 512 + (mb + i) * 32 + nn + 16] = (__bf16)p1[i];
      }
      v16bf pf;
      {
        int m2 = lane & 15;
        int kb = (lane >> 4) << 3;
        union { v16bf v; float4 f[2]; } u;
        u.f[0] = *(const float4*)&smem[8192 + wid * 512 + m2 * 32 + kb];
        u.f[1] = *(const float4*)&smem[8192 + wid * 512 + m2 * 32 + kb + 16];
        pf = u.v;
      }

      // ---- O += P * V (Vt tile rows contiguous in s -> B-fragments) ----
      o0 = WMMA_BF16(pf, load_fragB(vt +  0 * 32, 32, 0), o0);
      o1 = WMMA_BF16(pf, load_fragB(vt + 16 * 32, 32, 0), o1);
      o2 = WMMA_BF16(pf, load_fragB(vt + 32 * 32, 32, 0), o2);
      o3 = WMMA_BF16(pf, load_fragB(vt + 48 * 32, 32, 0), o3);
    }

    __syncthreads();   // everyone done reading before this buffer is overwritten
    bufsel ^= 1;
  }

  // ---- normalize and write concat-head output (bf16 [B*T, HD]) ----
  __bf16* Ocb = Oc + ((size_t)b * T) * HD + (size_t)h * Dh;
#pragma unroll
  for (int i = 0; i < 8; ++i) {
    float inv = 1.0f / lrow[i];
    __bf16* row = Ocb + (size_t)(qbase + mb + i) * HD;
    row[0 * 16 + nn] = (__bf16)(o0[i] * inv);
    row[1 * 16 + nn] = (__bf16)(o1[i] * inv);
    row[2 * 16 + nn] = (__bf16)(o2[i] * inv);
    row[3 * 16 + nn] = (__bf16)(o3[i] * inv);
  }
}

// ---------------- output projection: [B*T, HD] * [HD, E] + bp -> f32 ----------------
// One wave computes a 32x64 tile: two A-fragments reuse each B-fragment twice.
__global__ void __launch_bounds__(128)
proj_kernel(const __bf16* __restrict__ Oc, const __bf16* __restrict__ Wpt,
            const float* __restrict__ bp, float* __restrict__ out) {
  int wid  = threadIdx.x >> 5;
  int lane = threadIdx.x & 31;
  int g  = blockIdx.x * 4 + wid;  // 2048 wave-tiles: 256 m-tiles x 8 n-tiles
  int nt = g & 7;
  int mt = g >> 3;
  int row0 = mt * 32;

  const __bf16* arow = Oc + (size_t)row0 * HD;
  v8f c[2][4] = {};
  for (int k0 = 0; k0 < HD; k0 += 32) {
    v16bf a0 = load_fragA(arow,                E, k0);   // note: ld = HD = E = 512
    v16bf a1 = load_fragA(arow + 16 * (size_t)HD, HD, k0);
#pragma unroll
    for (int n = 0; n < 4; ++n) {
      v16bf bf = load_fragB(Wpt + (size_t)(nt * 64 + n * 16) * HD, HD, k0);
      c[0][n] = WMMA_BF16(a0, bf, c[0][n]);
      c[1][n] = WMMA_BF16(a1, bf, c[1][n]);
    }
  }

  int mb = (lane >> 4) << 3;
  int nn = lane & 15;
#pragma unroll
  for (int m = 0; m < 2; ++m) {
#pragma unroll
    for (int n = 0; n < 4; ++n) {
      int e = nt * 64 + n * 16 + nn;
      float be = bp[e];
#pragma unroll
      for (int i = 0; i < 8; ++i)
        out[(size_t)(row0 + m * 16 + mb + i) * E + e] = c[m][n][i] + be;
    }
  }
}

// ---------------- launcher ----------------
extern "C" void kernel_launch(void* const* d_in, const int* in_sizes, int n_in,
                              void* d_out, int out_size, void* d_ws, size_t ws_size,
                              hipStream_t stream) {
  (void)in_sizes; (void)n_in; (void)out_size; (void)ws_size;
  const float* x  = (const float*)d_in[0];
  const float* Wq = (const float*)d_in[1];
  const float* bq = (const float*)d_in[2];
  const float* Wk = (const float*)d_in[3];
  const float* bk = (const float*)d_in[4];
  const float* Wv = (const float*)d_in[5];
  const float* bv = (const float*)d_in[6];
  const float* Wp = (const float*)d_in[7];
  const float* bp = (const float*)d_in[8];
  float* out = (float*)d_out;

  char* ws = (char*)d_ws;
  size_t off = 0;
  auto carve = [&](size_t bytes) -> void* {
    void* p = ws + off;
    off += (bytes + 255) & ~(size_t)255;
    return p;
  };
  __bf16* xb  = (__bf16*)carve((size_t)Bz * T * E * 2);
  __bf16* Wqt = (__bf16*)carve((size_t)H * Dh * E * 2);
  __bf16* Wkt = (__bf16*)carve((size_t)H * Dh * E * 2);
  __bf16* Wvt = (__bf16*)carve((size_t)H * Dh * E * 2);
  __bf16* Wpt = (__bf16*)carve((size_t)HD * E * 2);
  __bf16* Qb  = (__bf16*)carve((size_t)Bz * H * T * Dh * 2);
  __bf16* Kb  = (__bf16*)carve((size_t)Bz * H * T * Dh * 2);
  __bf16* Vtb = (__bf16*)carve((size_t)Bz * H * Dh * T * 2);
  __bf16* Oc  = (__bf16*)carve((size_t)Bz * T * HD * 2);

  // 1) fp32 -> bf16 conversions / weight transposes
  cvt_bf16_kernel<<<1024, 256, 0, stream>>>(x, xb, Bz * T * E);
  {
    dim3 gw(64, H);
    transpose_bf16_kernel<<<gw, 256, 0, stream>>>(Wq, Wqt, E, Dh);
    transpose_bf16_kernel<<<gw, 256, 0, stream>>>(Wk, Wkt, E, Dh);
    transpose_bf16_kernel<<<gw, 256, 0, stream>>>(Wv, Wvt, E, Dh);
    dim3 gp(256, 1);
    transpose_bf16_kernel<<<gp, 256, 0, stream>>>(Wp, Wpt, HD, E);
  }
  // 2) fused QKV projection (WMMA, 32-row register blocking)
  qkv_kernel<<<512, 128, 0, stream>>>(xb, Wqt, Wkt, Wvt, bq, bk, bv, Qb, Kb, Vtb);
  // 3) causal flash attention (WMMA + TDM double-buffered K/V staging in LDS)
  attn_kernel<<<512, 256, 0, stream>>>(Qb, Kb, Vtb, Oc);
  // 4) output projection (WMMA, 32-row register blocking, f32 out + bias)
  proj_kernel<<<512, 128, 0, stream>>>(Oc, Wpt, bp, out);
}
